// WindowAttention_27384711479401
// MI455X (gfx1250) — matrix operand order repaired and checked by reading, not compiled, not run on hardware
//
#include <hip/hip_runtime.h>

// ---------------------------------------------------------------------------
// Fused Swin window attention for MI455X (gfx1250, wave32, WMMA bf16).
// One workgroup (4 waves = 4 heads) per 49-token window. All intermediates in
// LDS; only x is read and out written to HBM (~206 MB total -> ~9us floor).
// ---------------------------------------------------------------------------

typedef __attribute__((ext_vector_type(16))) __bf16 bf16x16;
typedef __attribute__((ext_vector_type(8)))  __bf16 bf16x8;
typedef __attribute__((ext_vector_type(8)))  float  f32x8;
typedef __attribute__((ext_vector_type(4)))  float  f32x4;

// LDS layout (units: bf16 halves). Strides are multiples of 8 halves (16B)
// so every b128 DS access is naturally aligned.
#define XS_RS 136   // x staging row stride (64 rows x 128 cols, padded)
#define QK_RS 40    // Q/K row stride  ([head][64 tok][32 d])
#define VT_RS 72    // V^T row stride  ([head][32 d][64 tok])
#define P_RS  72    // per-wave softmax-P bounce (16 x 64)
#define OS_RS 136   // O concat row stride (64 tok x 128 feat)

#define OFF_X 0
#define OFF_Q 8704     // 64*136
#define OFF_K 18944    // + 4*64*40
#define OFF_V 29184    // + 4*64*40
#define OFF_O 38400    // + 4*32*72
#define SMEM_HALVES 47104  // + 64*136  -> 94208 bytes (< 320KB/WGP, 3 WGs/WGP)
#define SMEM_BYTES (SMEM_HALVES * 2)

__device__ __forceinline__ f32x8 fzero8() {
  f32x8 z = {0.f, 0.f, 0.f, 0.f, 0.f, 0.f, 0.f, 0.f};
  return z;
}

// A operand (16x32 bf16) from an LDS row-major matrix.
// Lane l: row = row0 + l%16; K = k0 + (l/16)*16 .. +15.
__device__ __forceinline__ bf16x16 ld_a(const __bf16* base, int rs, int row0,
                                        int k0, int ln, int hh) {
  const __bf16* p = base + (row0 + ln) * rs + k0 + hh * 16;
  bf16x8 lo = *(const bf16x8*)(p);
  bf16x8 hi = *(const bf16x8*)(p + 8);
  return __builtin_shufflevector(lo, hi, 0, 1, 2, 3, 4, 5, 6, 7,
                                 8, 9, 10, 11, 12, 13, 14, 15);
}

// B operand (32x16 bf16). Source is stored "B-column-major": source row n is
// column n of B. Lane l: col = col0 + l%16, K = k0 + (l/16)*16 .. +15.
__device__ __forceinline__ bf16x16 ld_b(const __bf16* base, int rs, int col0,
                                        int k0, int ln, int hh) {
  const __bf16* p = base + (col0 + ln) * rs + k0 + hh * 16;
  bf16x8 lo = *(const bf16x8*)(p);
  bf16x8 hi = *(const bf16x8*)(p + 8);
  return __builtin_shufflevector(lo, hi, 0, 1, 2, 3, 4, 5, 6, 7,
                                 8, 9, 10, 11, 12, 13, 14, 15);
}

// B operand straight from a global f32 weight matrix stored [out_f][in_c]
// (row-major = B-column-major), converted to bf16 on the fly.
__device__ __forceinline__ bf16x16 ld_b_gf32(const float* w, int ld, int col0,
                                             int k0, int ln, int hh) {
  const float* p = w + (size_t)(col0 + ln) * ld + k0 + hh * 16;
  f32x4 v0 = *(const f32x4*)(p);
  f32x4 v1 = *(const f32x4*)(p + 4);
  f32x4 v2 = *(const f32x4*)(p + 8);
  f32x4 v3 = *(const f32x4*)(p + 12);
  bf16x16 b;
#pragma unroll
  for (int i = 0; i < 4; ++i) {
    b[i]      = (__bf16)v0[i];
    b[4 + i]  = (__bf16)v1[i];
    b[8 + i]  = (__bf16)v2[i];
    b[12 + i] = (__bf16)v3[i];
  }
  return b;
}

__device__ __forceinline__ f32x8 wmma_bf16(bf16x16 a, bf16x16 b, f32x8 c) {
  return __builtin_amdgcn_wmma_f32_16x16x32_bf16(false, a, false, b,
                                                 (short)0, c, false, false);
}

__global__ void __launch_bounds__(128)
win_attn_fused(const float* __restrict__ x,
               const float* __restrict__ qkv_w,   // [384][128]
               const float* __restrict__ qkv_b,   // [384]
               const float* __restrict__ proj_w,  // [128][128]
               const float* __restrict__ proj_b,  // [128]
               const float* __restrict__ rel_table, // [169][4]
               const int*   __restrict__ rel_index, // [49][49]
               float* __restrict__ out) {
  extern __shared__ __bf16 smem[];
  __bf16* xs = smem + OFF_X;
  __bf16* Qs = smem + OFF_Q;
  __bf16* Ks = smem + OFF_K;
  __bf16* Vt = smem + OFF_V;
  __bf16* Os = smem + OFF_O;

  const int win  = blockIdx.x;
  const int tid  = threadIdx.x;
  const int wave = tid >> 5;      // 4 waves = 4 heads
  const int lane = tid & 31;
  const int ln   = lane & 15;
  const int hh   = lane >> 4;

  // ---- Phase 0: stage x tile (49x128 f32 -> 64x128 bf16, zero padded) ----
  {
    const float* xw = x + (size_t)win * 49 * 128;
    for (int i = tid; i < 64 * 32; i += 128) {
      const int r = i >> 5;
      const int c = (i & 31) * 4;
      f32x4 v = {0.f, 0.f, 0.f, 0.f};
      if (r < 49) v = *(const f32x4*)(xw + r * 128 + c);
      __bf16* d = xs + r * XS_RS + c;
      d[0] = (__bf16)v[0]; d[1] = (__bf16)v[1];
      d[2] = (__bf16)v[2]; d[3] = (__bf16)v[3];
    }
  }
  __syncthreads();

  // ---- Phase 1: QKV = x @ qkv_w^T + b.  M=64, N=384, K=128. -------------
  // 24 N-tiles split across waves; epilogue scatters into head-major Q/K and
  // transposed V so attention operands come straight out of LDS.
  const float scale = 0.17677669529663687f;  // 1/sqrt(32), folded into Q
  for (int nt = wave; nt < 24; nt += 4) {
    const int f0 = nt * 16;
    f32x8 acc[4];
#pragma unroll
    for (int mt = 0; mt < 4; ++mt) acc[mt] = fzero8();
#pragma unroll
    for (int ks = 0; ks < 4; ++ks) {
      const int k0 = ks * 32;
      bf16x16 b = ld_b_gf32(qkv_w, 128, f0, k0, ln, hh);
#pragma unroll
      for (int mt = 0; mt < 4; ++mt) {
        bf16x16 a = ld_a(xs, XS_RS, mt * 16, k0, ln, hh);
        acc[mt] = wmma_bf16(a, b, acc[mt]);
      }
    }
    const int sel    = f0 >> 7;        // 0=Q 1=K 2=V
    const int head   = (f0 >> 5) & 3;
    const int d0     = f0 & 31;
    const float bias = qkv_b[f0 + ln];
    const float scl  = (sel == 0) ? scale : 1.0f;
#pragma unroll
    for (int mt = 0; mt < 4; ++mt) {
#pragma unroll
      for (int v = 0; v < 8; ++v) {
        const int m = mt * 16 + v + 8 * hh;        // token row (C layout)
        const float val = (acc[mt][v] + bias) * scl;
        if (sel == 0)      Qs[(head * 64 + m) * QK_RS + d0 + ln] = (__bf16)val;
        else if (sel == 1) Ks[(head * 64 + m) * QK_RS + d0 + ln] = (__bf16)val;
        else               Vt[(head * 32 + d0 + ln) * VT_RS + m] = (__bf16)val;
      }
    }
  }
  __syncthreads();

  // ---- Phase 2: per-head attention (wave == head). ----------------------
  // S = Q @ K^T (+bias), masked softmax over 49 valid cols, O = P @ V.
  {
    const int head   = wave;
    const __bf16* Qh = Qs + head * 64 * QK_RS;
    const __bf16* Kh = Ks + head * 64 * QK_RS;
    const __bf16* Vh = Vt + head * 32 * VT_RS;
    __bf16* Pw = xs + wave * 16 * P_RS;   // reuse x staging region per wave

    for (int mt = 0; mt < 4; ++mt) {
      bf16x16 aq = ld_a(Qh, QK_RS, mt * 16, 0, ln, hh);  // 16x32, K=hd=32
      f32x8 s[4];
#pragma unroll
      for (int jt = 0; jt < 4; ++jt) {
        bf16x16 b = ld_b(Kh, QK_RS, jt * 16, 0, ln, hh);
        s[jt] = wmma_bf16(aq, b, fzero8());
      }
      // Relative-position bias gather + padding mask.
#pragma unroll
      for (int jt = 0; jt < 4; ++jt) {
        const int j = jt * 16 + ln;
#pragma unroll
        for (int v = 0; v < 8; ++v) {
          const int m = mt * 16 + v + 8 * hh;
          if (j < 49) {
            float bb = 0.f;
            if (m < 49) bb = rel_table[rel_index[m * 49 + j] * 4 + head];
            s[jt][v] += bb;
          } else {
            s[jt][v] = -1e30f;   // mask padded key columns
          }
        }
      }
      // Softmax: each row lives in one 16-lane half-wave (C layout), so
      // xor-shuffles 8/4/2/1 reduce it without LDS.
#pragma unroll
      for (int v = 0; v < 8; ++v) {
        float mx = fmaxf(fmaxf(s[0][v], s[1][v]), fmaxf(s[2][v], s[3][v]));
#pragma unroll
        for (int off = 8; off > 0; off >>= 1)
          mx = fmaxf(mx, __shfl_xor(mx, off, 32));
        float sum = 0.f;
#pragma unroll
        for (int jt = 0; jt < 4; ++jt) {
          float e = __expf(s[jt][v] - mx);
          s[jt][v] = e;
          sum += e;
        }
#pragma unroll
        for (int off = 8; off > 0; off >>= 1)
          sum += __shfl_xor(sum, off, 32);
        const float inv = 1.0f / sum;
#pragma unroll
        for (int jt = 0; jt < 4; ++jt) s[jt][v] *= inv;
      }
      // Relayout P: C layout -> A layout via per-wave LDS bounce.
#pragma unroll
      for (int jt = 0; jt < 4; ++jt)
#pragma unroll
        for (int v = 0; v < 8; ++v)
          Pw[(v + 8 * hh) * P_RS + jt * 16 + ln] = (__bf16)s[jt][v];
      asm volatile("s_wait_dscnt 0x0" ::: "memory");  // same-wave LDS RAW

      // O(16x32) = P(16x64) @ V(64x32)
      f32x8 o0 = fzero8(), o1 = fzero8();
#pragma unroll
      for (int ks = 0; ks < 2; ++ks) {
        bf16x16 a = ld_a(Pw, P_RS, 0, ks * 32, ln, hh);
        o0 = wmma_bf16(a, ld_b(Vh, VT_RS, 0,  ks * 32, ln, hh), o0);
        o1 = wmma_bf16(a, ld_b(Vh, VT_RS, 16, ks * 32, ln, hh), o1);
      }
#pragma unroll
      for (int v = 0; v < 8; ++v) {
        const int m = mt * 16 + v + 8 * hh;
        Os[m * OS_RS + head * 32 + ln]      = (__bf16)o0[v];
        Os[m * OS_RS + head * 32 + 16 + ln] = (__bf16)o1[v];
      }
    }
  }
  __syncthreads();

  // ---- Phase 3: out = O @ proj_w^T + b.  M=64, N=128, K=128. ------------
  float* ow = out + (size_t)win * 49 * 128;
#pragma unroll
  for (int rep = 0; rep < 2; ++rep) {
    const int f0 = (wave + rep * 4) * 16;
    f32x8 acc[4];
#pragma unroll
    for (int mt = 0; mt < 4; ++mt) acc[mt] = fzero8();
#pragma unroll
    for (int ks = 0; ks < 4; ++ks) {
      bf16x16 b = ld_b_gf32(proj_w, 128, f0, ks * 32, ln, hh);
#pragma unroll
      for (int mt = 0; mt < 4; ++mt) {
        bf16x16 a = ld_a(Os, OS_RS, mt * 16, ks * 32, ln, hh);
        acc[mt] = wmma_bf16(a, b, acc[mt]);
      }
    }
    const float pb = proj_b[f0 + ln];
#pragma unroll
    for (int mt = 0; mt < 4; ++mt)
#pragma unroll
      for (int v = 0; v < 8; ++v) {
        const int m = mt * 16 + v + 8 * hh;
        if (m < 49) ow[m * 128 + f0 + ln] = acc[mt][v] + pb;
      }
  }
}

extern "C" void kernel_launch(void* const* d_in, const int* in_sizes, int n_in,
                              void* d_out, int out_size, void* d_ws,
                              size_t ws_size, hipStream_t stream) {
  const float* x       = (const float*)d_in[0];
  // d_in[1] = q_global (unused by forward)
  const float* qkv_w   = (const float*)d_in[2];
  const float* qkv_b   = (const float*)d_in[3];
  const float* proj_w  = (const float*)d_in[4];
  const float* proj_b  = (const float*)d_in[5];
  const float* rel_tab = (const float*)d_in[6];
  const int*   rel_idx = (const int*)d_in[7];
  float* out = (float*)d_out;
  (void)n_in; (void)d_ws; (void)ws_size; (void)out_size;

  const int num_windows = in_sizes[0] / (49 * 128);  // 4096
  hipLaunchKernelGGL(win_attn_fused, dim3(num_windows), dim3(128), SMEM_BYTES,
                     stream, x, qkv_w, qkv_b, proj_w, proj_b, rel_tab, rel_idx,
                     out);
}
